// BNN_3blk_200_100_47201690583464
// MI455X (gfx1250) — 16-bit and 8-bit matrix operand order repaired and checked by reading, compiled, not run on hardware
//
#include <hip/hip_runtime.h>
#include <math.h>

typedef __attribute__((ext_vector_type(16))) _Float16 v16h;
typedef __attribute__((ext_vector_type(8)))  float    v8f;
typedef __attribute__((ext_vector_type(8)))  int      v8i;
typedef __attribute__((ext_vector_type(4)))  float    vf4;

#define NROWS 65536
#define DIN   1024
#define N1T   13
#define N1P   208
#define N1A   200
#define NBLK  256
#define RPB   256

#define OFF_W1H  0u
#define OFF_W2I  425984u
#define OFF_W3I  454656u
#define OFF_W5I  468992u
#define OFF_SS1  471040u
#define OFF_SS2  473088u
#define OFF_SS3  475136u
#define OFF_PART 477184u
#define OFF_Y1   1001472u
#define OFF_Y2   55527424u
#define OFF_Y3   84887552u

__global__ void k_prep_w1(const float* __restrict__ w, _Float16* __restrict__ wh)
{
    int idx = blockIdx.x * 256 + threadIdx.x;
    if (idx >= N1P * DIN) return;
    int n = idx / DIN;
    _Float16 o = (_Float16)0.f;
    if (n < N1A) o = (w[idx] >= 0.f) ? (_Float16)1.f : (_Float16)(-1.f);
    wh[idx] = o;
}

__global__ void k_prep_wi8(const float* __restrict__ w, signed char* __restrict__ wi,
                           int nact, int Ksrc, int Kdst, int ntot)
{
    int idx = blockIdx.x * 256 + threadIdx.x;
    if (idx >= ntot) return;
    int n = idx / Kdst;
    int k = idx - n * Kdst;
    signed char o = 0;
    if (n < nact && k < Ksrc)
        o = (w[n * Ksrc + k] >= 0.f) ? (signed char)1 : (signed char)(-1);
    wi[idx] = o;
}

__global__ __launch_bounds__(256) void k_gemm1(const float* __restrict__ x,
                                               const _Float16* __restrict__ w1h,
                                               float* __restrict__ y1)
{
    const int lane  = threadIdx.x & 31;
    const int wave  = threadIdx.x >> 5;
    const int l16   = lane & 15;
    const int hi    = lane >> 4;
    const int sel8  = hi << 3;
    const int sel16 = hi << 4;
    const int m0    = (blockIdx.x * 8 + wave) * 16;

    const float* xr = x + (size_t)(m0 + l16) * DIN;

    v8f acc[N1T];
#pragma unroll
    for (int t = 0; t < N1T; ++t) { v8f z = {}; acc[t] = z; }

    for (int k0 = 0; k0 < DIN; k0 += 32) {
        vf4 q0 = *(const vf4*)(xr + k0 + sel16);
        vf4 q1 = *(const vf4*)(xr + k0 + sel16 + 4);
        vf4 q2 = *(const vf4*)(xr + k0 + sel16 + 8);
        vf4 q3 = *(const vf4*)(xr + k0 + sel16 + 12);
        v16h ahi, alo;
#pragma unroll
        for (int i = 0; i < 4; ++i) {
            float v;
            v = q0[i]; { _Float16 h = (_Float16)v; ahi[i]    = h; alo[i]    = (_Float16)(v - (float)h); }
            v = q1[i]; { _Float16 h = (_Float16)v; ahi[4+i]  = h; alo[4+i]  = (_Float16)(v - (float)h); }
            v = q2[i]; { _Float16 h = (_Float16)v; ahi[8+i]  = h; alo[8+i]  = (_Float16)(v - (float)h); }
            v = q3[i]; { _Float16 h = (_Float16)v; ahi[12+i] = h; alo[12+i] = (_Float16)(v - (float)h); }
        }
#pragma unroll
        for (int t = 0; t < N1T; ++t) {
            v16h b = *(const v16h*)(w1h + (size_t)(t*16 + l16) * DIN + k0 + sel16);
            acc[t] = __builtin_amdgcn_wmma_f32_16x16x32_f16(false, ahi, false, b, (short)0, acc[t], false, false);
            acc[t] = __builtin_amdgcn_wmma_f32_16x16x32_f16(false, alo, false, b, (short)0, acc[t], false, false);
        }
    }
    const int rbase = m0 + hi * 8;
#pragma unroll
    for (int t = 0; t < N1T; ++t) {
        int col = t*16 + l16;
#pragma unroll
        for (int r = 0; r < 8; ++r)
            y1[(size_t)(rbase + r) * N1P + col] = acc[t][r];
    }
}

__global__ __launch_bounds__(256) void k_stats_part(const float* __restrict__ y,
                                                    int stride, int ncols,
                                                    float* __restrict__ part)
{
    int c = threadIdx.x;
    size_t r0 = (size_t)blockIdx.x * RPB;
    float s = 0.f, s2 = 0.f;
    if (c < ncols) {
        for (int r = 0; r < RPB; ++r) {
            float v = y[(r0 + r) * stride + c];
            s += v; s2 += v * v;
        }
    }
    part[blockIdx.x * 512 + c]       = s;
    part[blockIdx.x * 512 + 256 + c] = s2;
}

__global__ void k_stats_final(const float* __restrict__ part,
                              const float* __restrict__ g, const float* __restrict__ be,
                              int ncols, float* __restrict__ ss)
{
    int c = threadIdx.x;
    if (c >= ncols) return;
    double s = 0.0, s2 = 0.0;
    for (int b = 0; b < NBLK; ++b) {
        s  += (double)part[b * 512 + c];
        s2 += (double)part[b * 512 + 256 + c];
    }
    double mu  = s  / (double)NROWS;
    double var = s2 / (double)NROWS - mu * mu;
    float scale = g[c] * rsqrtf((float)var + 1e-4f);
    float shift = be[c] - (float)mu * scale;
    ss[c]       = scale;
    ss[256 + c] = shift;
}

template<int NT, int KP>
__global__ __launch_bounds__(256) void k_bingemm(
    const float* __restrict__ yIn, int strideIn, int KA,
    const float* __restrict__ ss,
    const signed char* __restrict__ w,
    float* __restrict__ yOut, int strideOut)
{
    const int lane  = threadIdx.x & 31;
    const int wave  = threadIdx.x >> 5;
    const int l16   = lane & 15;
    const int hi    = lane >> 4;
    const int sel8  = hi << 3;
    const int sel16 = hi << 4;
    const int m0    = (blockIdx.x * 8 + wave) * 16;
    const float* yr = yIn + (size_t)(m0 + l16) * strideIn;

    v8i acc[NT];
#pragma unroll
    for (int t = 0; t < NT; ++t) { v8i z = {}; acc[t] = z; }

    for (int k0 = 0; k0 < KP; k0 += 64) {
        v8i a;
#pragma unroll
        for (int j = 0; j < 4; ++j) {
            int kc = k0 + j*16 + sel8;
            unsigned d0 = 0u, d1 = 0u;
            if (kc + 8 <= KA) {
                vf4 v0 = *(const vf4*)(yr + kc);
                vf4 v1 = *(const vf4*)(yr + kc + 4);
#pragma unroll
                for (int i = 0; i < 4; ++i) {
                    float bn0 = v0[i] * ss[kc+i]   + ss[256+kc+i];
                    float bn1 = v1[i] * ss[kc+4+i] + ss[256+kc+4+i];
                    d0 |= (bn0 >= 0.f ? 0x01u : 0xFFu) << (8*i);
                    d1 |= (bn1 >= 0.f ? 0x01u : 0xFFu) << (8*i);
                }
            } else {
#pragma unroll
                for (int i = 0; i < 8; ++i) {
                    int k = kc + i;
                    unsigned by = 0u;
                    if (k < KA) {
                        float bn = yr[k] * ss[k] + ss[256+k];
                        by = (bn >= 0.f) ? 0x01u : 0xFFu;
                    }
                    if (i < 4) d0 |= by << (8*i); else d1 |= by << (8*(i-4));
                }
            }
            a[2*j]   = (int)d0;
            a[2*j+1] = (int)d1;
        }
#pragma unroll
        for (int t = 0; t < NT; ++t) {
            const signed char* wr = w + (size_t)(t*16 + l16) * KP + k0 + sel8;
            const int* p0 = (const int*)wr;
            const int* p1 = (const int*)(wr + 32);
            v8i b;
            b[0]=p0[0]; b[1]=p0[1]; b[2]=p0[4]; b[3]=p0[5];
            b[4]=p1[0]; b[5]=p1[1]; b[6]=p1[4]; b[7]=p1[5];
            acc[t] = __builtin_amdgcn_wmma_i32_16x16x64_iu8(true, a, true, b, acc[t], false, false);
        }
    }
    const int rbase = m0 + hi * 8;
#pragma unroll
    for (int t = 0; t < NT; ++t) {
        int col = t*16 + l16;
#pragma unroll
        for (int r = 0; r < 8; ++r)
            yOut[(size_t)(rbase + r) * strideOut + col] = (float)acc[t][r];
    }
}

__global__ __launch_bounds__(256) void k_final(
    const float* __restrict__ yIn, int strideIn, int KA,
    const float* __restrict__ ss,
    const signed char* __restrict__ w5i,
    const float* __restrict__ b5,
    float* __restrict__ out)
{
    const int lane  = threadIdx.x & 31;
    const int wave  = threadIdx.x >> 5;
    const int l16   = lane & 15;
    const int hi    = lane >> 4;
    const int sel8  = hi << 3;
    const int sel16 = hi << 4;
    const int m0    = (blockIdx.x * 8 + wave) * 16;
    const float* yr = yIn + (size_t)(m0 + l16) * strideIn;

    v8i acc = {};
    for (int k0 = 0; k0 < 128; k0 += 64) {
        v8i a;
#pragma unroll
        for (int j = 0; j < 4; ++j) {
            int kc = k0 + j*16 + sel8;
            unsigned d0 = 0u, d1 = 0u;
            if (kc + 8 <= KA) {
                vf4 v0 = *(const vf4*)(yr + kc);
                vf4 v1 = *(const vf4*)(yr + kc + 4);
#pragma unroll
                for (int i = 0; i < 4; ++i) {
                    float bn0 = v0[i] * ss[kc+i]   + ss[256+kc+i];
                    float bn1 = v1[i] * ss[kc+4+i] + ss[256+kc+4+i];
                    d0 |= (bn0 >= 0.f ? 0x01u : 0xFFu) << (8*i);
                    d1 |= (bn1 >= 0.f ? 0x01u : 0xFFu) << (8*i);
                }
            } else {
#pragma unroll
                for (int i = 0; i < 8; ++i) {
                    int k = kc + i;
                    unsigned by = 0u;
                    if (k < KA) {
                        float bn = yr[k] * ss[k] + ss[256+k];
                        by = (bn >= 0.f) ? 0x01u : 0xFFu;
                    }
                    if (i < 4) d0 |= by << (8*i); else d1 |= by << (8*(i-4));
                }
            }
            a[2*j]   = (int)d0;
            a[2*j+1] = (int)d1;
        }
        const signed char* wr = w5i + (size_t)l16 * 128 + k0 + sel8;
        const int* p0 = (const int*)wr;
        const int* p1 = (const int*)(wr + 32);
        v8i b;
        b[0]=p0[0]; b[1]=p0[1]; b[2]=p0[4]; b[3]=p0[5];
        b[4]=p1[0]; b[5]=p1[1]; b[6]=p1[4]; b[7]=p1[5];
        acc = __builtin_amdgcn_wmma_i32_16x16x64_iu8(true, a, true, b, acc, false, false);
    }

    const int col    = l16;
    const bool active = (col < 10);
    const float bias = active ? b5[col] : 0.f;
    const int rbase  = m0 + hi * 8;
#pragma unroll
    for (int r = 0; r < 8; ++r) {
        float logit = (float)acc[r] + bias;
        float lm = active ? logit : -1e30f;
#pragma unroll
        for (int m = 1; m < 16; m <<= 1) lm = fmaxf(lm, __shfl_xor(lm, m, 32));
        float e = active ? expf(logit - lm) : 0.f;
#pragma unroll
        for (int m = 1; m < 16; m <<= 1) e += __shfl_xor(e, m, 32);
        float res = logit - lm - logf(e);
        if (active) out[(size_t)(rbase + r) * 10 + col] = res;
    }
}

extern "C" void kernel_launch(void* const* d_in, const int* in_sizes, int n_in,
                              void* d_out, int out_size, void* d_ws, size_t ws_size,
                              hipStream_t stream)
{
    (void)in_sizes; (void)n_in; (void)out_size; (void)ws_size;
    const float* x   = (const float*)d_in[0];
    const float* w1  = (const float*)d_in[1];
    const float* g1  = (const float*)d_in[3];
    const float* be1 = (const float*)d_in[4];
    const float* w2  = (const float*)d_in[5];
    const float* g2  = (const float*)d_in[7];
    const float* be2 = (const float*)d_in[8];
    const float* w3  = (const float*)d_in[9];
    const float* g3  = (const float*)d_in[11];
    const float* be3 = (const float*)d_in[12];
    const float* w5  = (const float*)d_in[13];
    const float* b5  = (const float*)d_in[14];
    float* out = (float*)d_out;
    char* ws = (char*)d_ws;

    _Float16*    w1h = (_Float16*)(ws + OFF_W1H);
    signed char* w2i = (signed char*)(ws + OFF_W2I);
    signed char* w3i = (signed char*)(ws + OFF_W3I);
    signed char* w5i = (signed char*)(ws + OFF_W5I);
    float* ss1  = (float*)(ws + OFF_SS1);
    float* ss2  = (float*)(ws + OFF_SS2);
    float* ss3  = (float*)(ws + OFF_SS3);
    float* part = (float*)(ws + OFF_PART);
    float* y1   = (float*)(ws + OFF_Y1);
    float* y2   = (float*)(ws + OFF_Y2);
    float* y3   = (float*)(ws + OFF_Y3);

    k_prep_w1 <<<(N1P*DIN + 255)/256, 256, 0, stream>>>(w1, w1h);
    k_prep_wi8<<<(112*256 + 255)/256, 256, 0, stream>>>(w2, w2i, 100, 200, 256, 112*256);
    k_prep_wi8<<<(112*128 + 255)/256, 256, 0, stream>>>(w3, w3i, 100, 100, 128, 112*128);
    k_prep_wi8<<<( 16*128 + 255)/256, 256, 0, stream>>>(w5, w5i,  10, 100, 128,  16*128);

    k_gemm1<<<512, 256, 0, stream>>>(x, w1h, y1);
    k_stats_part<<<NBLK, 256, 0, stream>>>(y1, N1P, N1A, part);
    k_stats_final<<<1, 256, 0, stream>>>(part, g1, be1, N1A, ss1);

    k_bingemm<7, 256><<<512, 256, 0, stream>>>(y1, N1P, N1A, ss1, w2i, y2, 112);
    k_stats_part<<<NBLK, 256, 0, stream>>>(y2, 112, 100, part);
    k_stats_final<<<1, 256, 0, stream>>>(part, g2, be2, 100, ss2);

    k_bingemm<7, 128><<<512, 256, 0, stream>>>(y2, 112, 100, ss2, w3i, y3, 112);
    k_stats_part<<<NBLK, 256, 0, stream>>>(y3, 112, 100, part);
    k_stats_final<<<1, 256, 0, stream>>>(part, g3, be3, 100, ss3);

    k_final<<<512, 256, 0, stream>>>(y3, 112, 100, ss3, w5i, b5, out);
}
